// Attention2_2327872274830
// MI455X (gfx1250) — matrix operand order repaired and checked by reading, not compiled, not run on hardware
//
#include <hip/hip_runtime.h>
#include <hip/hip_bf16.h>

// MI455X (gfx1250) implementation. wave32, WMMA f32<-f16 16x16x32.
// Pipeline of kernels; BN stats via two-pass (block LDS reduce + global atomics).
// Tile kernels: 16 points per wave, FULL/tail template split to keep the hot
// path free of per-element bounds cndmasks; gathers use b128 vector loads.

typedef __attribute__((ext_vector_type(16))) _Float16 v16h;
typedef __attribute__((ext_vector_type(8)))  float    v8f;

union AFrag { v16h v; _Float16 h[16]; };
union DFrag { v8f  v; float    f[8];  };
union Row20 { float4 q[5]; float f[20]; };   // one 20-channel row (80B, 16B aligned)

#define KNN     9
#define C_IN    20
#define C_INNER 20
#define BN_EPS  1e-5f

// A-fragment half index -> K (16-bit A 16x32 layout, ISA 7.12.2)
__device__ __forceinline__ int akmap(int i, int half) {
  return (i < 8) ? (8 * half + i) : (8 * half + 8 + i);
}

// ---- W1 (20 x 23, row major) B-fragment for output tile ntile (cols 16*ntile..)
__device__ __forceinline__ v16h w1_frag(const float* __restrict__ W1, int ntile) {
  const int lane = threadIdx.x & 31;
  const int half = lane >> 4;
  const int n = (lane & 15) + 16 * ntile;   // output channel (B column)
  AFrag b;
#pragma unroll
  for (int i = 0; i < 16; ++i) {
    const int K = akmap(i, half);           // input channel (B row)
    float v = 0.f;
    if (n < C_INNER && K < C_INNER + 3) v = W1[n * (C_INNER + 3) + K];
    b.h[i] = (_Float16)v;
  }
  return b.v;
}

// ---- conv tile: y[16 points x 20 ch] for neighbor slot k via 2 WMMAs.
// Row M = point; input row = [feature[idx] (20), points[idx]-points[p] (3), pad].
// Vectorized: half0 needs K 0..7,16..22 ; half1 needs K 8..15 (24..31 are pad).
__device__ __forceinline__ void conv_tile(
    const float* __restrict__ points, const float* __restrict__ feature,
    const int* __restrict__ index, v16h bt0, v16h bt1,
    int p, int k, float (&yA)[8], float (&yB)[8])
{
  const int lane = threadIdx.x & 31;
  const int half = lane >> 4;
  const int nb = index[p * KNN + k];
  const float* frow = feature + (long long)nb * C_INNER;
  const float4 r0 = *(const float4*)(frow + 8 * half);      // K 0..3  / 8..11
  const float4 r1 = *(const float4*)(frow + 8 * half + 4);  // K 4..7  / 12..15
  AFrag a;
  a.h[0] = (_Float16)r0.x; a.h[1] = (_Float16)r0.y;
  a.h[2] = (_Float16)r0.z; a.h[3] = (_Float16)r0.w;
  a.h[4] = (_Float16)r1.x; a.h[5] = (_Float16)r1.y;
  a.h[6] = (_Float16)r1.z; a.h[7] = (_Float16)r1.w;
  float tail[8];
#pragma unroll
  for (int j = 0; j < 8; ++j) tail[j] = 0.f;
  if (half == 0) {                                          // K 16..23
    const float4 r2 = *(const float4*)(frow + 16);
    tail[0] = r2.x; tail[1] = r2.y; tail[2] = r2.z; tail[3] = r2.w;
    tail[4] = points[nb * 3 + 0] - points[p * 3 + 0];
    tail[5] = points[nb * 3 + 1] - points[p * 3 + 1];
    tail[6] = points[nb * 3 + 2] - points[p * 3 + 2];
  }
#pragma unroll
  for (int j = 0; j < 8; ++j) a.h[8 + j] = (_Float16)tail[j];
  const v8f z = {};
  DFrag dA, dB;
  dA.v = __builtin_amdgcn_wmma_f32_16x16x32_f16(false, a.v, false, bt0, (short)0, z, false, false);
  dB.v = __builtin_amdgcn_wmma_f32_16x16x32_f16(false, a.v, false, bt1, (short)0, z, false, false);
#pragma unroll
  for (int v = 0; v < 8; ++v) { yA[v] = dA.f[v]; yB[v] = dB.f[v]; }
}

// ============ kernel 0: zero the stats region of ws ============
__global__ void k_zero(float* ws) {
  const int i = blockIdx.x * blockDim.x + threadIdx.x;
  if (i < 512) ws[i] = 0.f;
}

// ============ kernel 1: conv over all (n,k), accumulate BN1 sum/sumsq ============
template<bool FULL>
__device__ __forceinline__ void conv_stats_tile(
    const float* __restrict__ points, const float* __restrict__ feature,
    const int* __restrict__ index, v16h b0, v16h b1, int pbase, int N,
    float& sA, float& qA, float& sB, float& qB)
{
  const int lane = threadIdx.x & 31;
  const int half = lane >> 4;
  int p = pbase + (lane & 15);
  if (!FULL && p >= N) p = N - 1;
  for (int k = 0; k < KNN; ++k) {
    float yA[8], yB[8];
    conv_tile(points, feature, index, b0, b1, p, k, yA, yB);
#pragma unroll
    for (int v = 0; v < 8; ++v) {
      if (FULL || (pbase + v + 8 * half) < N) {
        sA += yA[v]; qA += yA[v] * yA[v];
        sB += yB[v]; qB += yB[v] * yB[v];
      }
    }
  }
}

__global__ __launch_bounds__(256) void k_conv_stats(
    const float* __restrict__ points, const float* __restrict__ feature,
    const int* __restrict__ index, const float* __restrict__ W1,
    float* __restrict__ stats, int N)
{
  __shared__ float lsum[32], lsq[32];
  if (threadIdx.x < 32) { lsum[threadIdx.x] = 0.f; lsq[threadIdx.x] = 0.f; }
  __syncthreads();
  const v16h b0 = w1_frag(W1, 0), b1 = w1_frag(W1, 1);
  const int lane = threadIdx.x & 31;
  const int wid = (blockIdx.x * blockDim.x + threadIdx.x) >> 5;
  const int nw  = (gridDim.x * blockDim.x) >> 5;
  const int ntiles = (N + 15) >> 4;
  float sA = 0.f, qA = 0.f, sB = 0.f, qB = 0.f;
  for (int t = wid; t < ntiles; t += nw) {
    if (t + nw < ntiles) {  // prefetch next tile's index rows into L2/L0
      const int pn = ((t + nw) << 4) + (lane & 15);
      __builtin_prefetch(index + pn * KNN, 0, 1);
    }
    const int pbase = t << 4;
    if (pbase + 16 <= N)
      conv_stats_tile<true >(points, feature, index, b0, b1, pbase, N, sA, qA, sB, qB);
    else
      conv_stats_tile<false>(points, feature, index, b0, b1, pbase, N, sA, qA, sB, qB);
  }
  const int cA = lane & 15, cB = 16 + (lane & 15);
  atomicAdd(&lsum[cA], sA); atomicAdd(&lsq[cA], qA);
  if (cB < C_INNER) { atomicAdd(&lsum[cB], sB); atomicAdd(&lsq[cB], qB); }
  __syncthreads();
  if (threadIdx.x < C_INNER) {
    atomicAdd(&stats[threadIdx.x], lsum[threadIdx.x]);
    atomicAdd(&stats[C_INNER + threadIdx.x], lsq[threadIdx.x]);
  }
}

// ============ finalize: sums -> per-channel affine (scale s, shift t) ============
__global__ void k_finalize(const float* __restrict__ acc,
                           const float* __restrict__ gamma, const float* __restrict__ beta,
                           float count, float* __restrict__ st)
{
  const int c = threadIdx.x;
  if (c < 20) {
    const float m = acc[c] / count;
    float var = acc[20 + c] / count - m * m;
    if (var < 0.f) var = 0.f;
    const float s = gamma[c] * rsqrtf(var + BN_EPS);
    st[c] = s;
    st[20 + c] = beta[c] - m * s;
  }
}

// ============ kernel 2: conv + BN1 + attention weights + weighted sum ============
template<bool FULL>
__device__ __forceinline__ void conv_attn_tile(
    const float* __restrict__ points, const float* __restrict__ feature,
    const int* __restrict__ index, v16h b0, v16h b1,
    float s1a, float t1a, float s1b, float t1b,
    int pbase, int N, float* __restrict__ wbuf, float* __restrict__ x1)
{
  const int lane = threadIdx.x & 31;
  const int half = lane >> 4;
  const int cA = lane & 15, cB = 16 + (lane & 15);
  int p = pbase + (lane & 15);
  if (!FULL && p >= N) p = N - 1;
  float y0A[8], y0B[8];
  conv_tile(points, feature, index, b0, b1, p, 0, y0A, y0B);
#pragma unroll
  for (int v = 0; v < 8; ++v) { y0A[v] = y0A[v] * s1a + t1a; y0B[v] = y0B[v] * s1b + t1b; }
  float xA[8], xB[8];
#pragma unroll
  for (int v = 0; v < 8; ++v) { xA[v] = 0.f; xB[v] = 0.f; }
  for (int k = 0; k < KNN; ++k) {
    float yA[8], yB[8];
    if (k == 0) {
#pragma unroll
      for (int v = 0; v < 8; ++v) { yA[v] = y0A[v]; yB[v] = y0B[v]; }
    } else {
      conv_tile(points, feature, index, b0, b1, p, k, yA, yB);
#pragma unroll
      for (int v = 0; v < 8; ++v) { yA[v] = yA[v] * s1a + t1a; yB[v] = yB[v] * s1b + t1b; }
    }
    // w[m,k] = sum over channels (lanes of the half) of y_k * y_0
#pragma unroll
    for (int v = 0; v < 8; ++v) {
      float pv = yA[v] * y0A[v] + yB[v] * y0B[v];
      for (int m = 1; m < 16; m <<= 1) pv += __shfl_xor(pv, m, 32);
      xA[v] += pv * yA[v];
      xB[v] += pv * yB[v];
      const int pm = pbase + v + 8 * half;
      if ((lane & 15) == v && (FULL || pm < N)) wbuf[pm * KNN + k] = pv;
    }
  }
#pragma unroll
  for (int v = 0; v < 8; ++v) {
    const int pm = pbase + v + 8 * half;
    if (FULL || pm < N) {
      x1[pm * C_INNER + cA] = xA[v];
      if (cB < C_INNER) x1[pm * C_INNER + cB] = xB[v];
    }
  }
}

__global__ __launch_bounds__(256) void k_conv_attn(
    const float* __restrict__ points, const float* __restrict__ feature,
    const int* __restrict__ index, const float* __restrict__ W1,
    const float* __restrict__ st1,
    float* __restrict__ wbuf, float* __restrict__ x1, int N)
{
  const v16h b0 = w1_frag(W1, 0), b1 = w1_frag(W1, 1);
  const int lane = threadIdx.x & 31;
  const int cA = lane & 15, cB = 16 + (lane & 15);
  const float s1a = st1[cA], t1a = st1[20 + cA];
  const float s1b = (cB < C_INNER) ? st1[cB] : 0.f;
  const float t1b = (cB < C_INNER) ? st1[20 + cB] : 0.f;
  const int wid = (blockIdx.x * blockDim.x + threadIdx.x) >> 5;
  const int nw  = (gridDim.x * blockDim.x) >> 5;
  const int ntiles = (N + 15) >> 4;
  for (int t = wid; t < ntiles; t += nw) {
    if (t + nw < ntiles) {
      const int pn = ((t + nw) << 4) + (lane & 15);
      __builtin_prefetch(index + pn * KNN, 0, 1);
    }
    const int pbase = t << 4;
    if (pbase + 16 <= N)
      conv_attn_tile<true >(points, feature, index, b0, b1, s1a, t1a, s1b, t1b, pbase, N, wbuf, x1);
    else
      conv_attn_tile<false>(points, feature, index, b0, b1, s1a, t1a, s1b, t1b, pbase, N, wbuf, x1);
  }
}

// ============ kernel 3: x2[n] = sum_k x1[index[n,k]] * w[n,k]; BN2 stats ============
__global__ __launch_bounds__(256) void k_propagate(
    const int* __restrict__ index, const float* __restrict__ wbuf,
    const float* __restrict__ x1, float* __restrict__ x2,
    float* __restrict__ stats2, int N)
{
  __shared__ float lsum[C_INNER], lsq[C_INNER];
  if (threadIdx.x < C_INNER) { lsum[threadIdx.x] = 0.f; lsq[threadIdx.x] = 0.f; }
  __syncthreads();
  float tsum[C_INNER], tsq[C_INNER];
#pragma unroll
  for (int c = 0; c < C_INNER; ++c) { tsum[c] = 0.f; tsq[c] = 0.f; }
  const int stride = gridDim.x * blockDim.x;
  for (int n = blockIdx.x * blockDim.x + threadIdx.x; n < N; n += stride) {
    if (n + stride < N) __builtin_prefetch(index + (n + stride) * KNN, 0, 1);
    float acc[C_INNER];
#pragma unroll
    for (int c = 0; c < C_INNER; ++c) acc[c] = 0.f;
    for (int k = 0; k < KNN; ++k) {
      const float wk = wbuf[n * KNN + k];
      const Row20 r = *(const Row20*)(x1 + (long long)index[n * KNN + k] * C_INNER);
#pragma unroll
      for (int c = 0; c < C_INNER; ++c) acc[c] += wk * r.f[c];
    }
    Row20 o;
#pragma unroll
    for (int c = 0; c < C_INNER; ++c) {
      o.f[c] = acc[c];
      tsum[c] += acc[c]; tsq[c] += acc[c] * acc[c];
    }
    *(Row20*)(x2 + n * C_INNER) = o;
  }
#pragma unroll
  for (int c = 0; c < C_INNER; ++c) { atomicAdd(&lsum[c], tsum[c]); atomicAdd(&lsq[c], tsq[c]); }
  __syncthreads();
  if (threadIdx.x < C_INNER) {
    atomicAdd(&stats2[threadIdx.x], lsum[threadIdx.x]);
    atomicAdd(&stats2[C_INNER + threadIdx.x], lsq[threadIdx.x]);
  }
}

// ============ kernel 4: u = concat(relu(bn2(x2)), feature) @ Wr1^T + br1; BN3 stats ======
template<bool FULL>
__device__ __forceinline__ void refine_tile(
    const float* __restrict__ feature, const float* __restrict__ x2,
    const float* __restrict__ st2, const v16h (&bf)[2][2],
    float brA, float brB, int pbase, int N, float* __restrict__ u,
    float& sA, float& qA, float& sB, float& qB)
{
  const int lane = threadIdx.x & 31;
  const int half = lane >> 4;
  const int cA = lane & 15, cB = 16 + (lane & 15);
  int p = pbase + (lane & 15);
  if (!FULL && p >= N) p = N - 1;
  const Row20 rx = *(const Row20*)(x2 + (long long)p * C_INNER);
  const Row20 rf = *(const Row20*)(feature + (long long)p * C_INNER);
  float z[C_INNER + C_IN];
#pragma unroll
  for (int c = 0; c < C_INNER; ++c) z[c] = fmaxf(rx.f[c] * st2[c] + st2[20 + c], 0.f);
#pragma unroll
  for (int c = 0; c < C_IN; ++c) z[C_INNER + c] = rf.f[c];
  v8f accA = {}, accB = {};
#pragma unroll
  for (int kt = 0; kt < 2; ++kt) {
    AFrag a;
#pragma unroll
    for (int i = 0; i < 16; ++i) {
      const int K = kt * 32 + akmap(i, half);
      a.h[i] = (K < C_INNER + C_IN) ? (_Float16)z[K] : (_Float16)0.f;
    }
    accA = __builtin_amdgcn_wmma_f32_16x16x32_f16(false, a.v, false, bf[kt][0], (short)0, accA, false, false);
    accB = __builtin_amdgcn_wmma_f32_16x16x32_f16(false, a.v, false, bf[kt][1], (short)0, accB, false, false);
  }
  DFrag dA, dB; dA.v = accA; dB.v = accB;
#pragma unroll
  for (int v = 0; v < 8; ++v) {
    const int pm = pbase + v + 8 * half;
    if (FULL || pm < N) {
      const float uA = dA.f[v] + brA;
      const float uB = dB.f[v] + brB;
      u[pm * C_IN + cA] = uA;
      if (cB < C_IN) u[pm * C_IN + cB] = uB;
      sA += uA; qA += uA * uA;
      sB += uB; qB += uB * uB;
    }
  }
}

__global__ __launch_bounds__(256) void k_refine1(
    const float* __restrict__ feature, const float* __restrict__ x2,
    const float* __restrict__ st2, const float* __restrict__ Wr1,
    const float* __restrict__ br1,
    float* __restrict__ u, float* __restrict__ stats3, int N)
{
  __shared__ float lsum[32], lsq[32];
  if (threadIdx.x < 32) { lsum[threadIdx.x] = 0.f; lsq[threadIdx.x] = 0.f; }
  __syncthreads();
  const int lane = threadIdx.x & 31;
  const int half = lane >> 4;
  const int cA = lane & 15, cB = 16 + (lane & 15);
  // Wr1 is (20 x 40) row major; B[k=c][n=o] = Wr1[o*40 + c]
  v16h bf[2][2];
#pragma unroll
  for (int kt = 0; kt < 2; ++kt)
#pragma unroll
    for (int nt = 0; nt < 2; ++nt) {
      AFrag b;
#pragma unroll
      for (int i = 0; i < 16; ++i) {
        const int K = kt * 32 + akmap(i, half);
        const int n = (lane & 15) + 16 * nt;
        float v = 0.f;
        if (n < C_IN && K < C_IN + C_INNER) v = Wr1[n * (C_IN + C_INNER) + K];
        b.h[i] = (_Float16)v;
      }
      bf[kt][nt] = b.v;
    }
  const float brA = br1[cA];
  const float brB = (cB < C_IN) ? br1[cB] : 0.f;
  float sA = 0.f, qA = 0.f, sB = 0.f, qB = 0.f;
  const int wid = (blockIdx.x * blockDim.x + threadIdx.x) >> 5;
  const int nw  = (gridDim.x * blockDim.x) >> 5;
  const int ntiles = (N + 15) >> 4;
  for (int t = wid; t < ntiles; t += nw) {
    if (t + nw < ntiles) {
      const int pn = ((t + nw) << 4) + (lane & 15);
      __builtin_prefetch(x2 + (long long)pn * C_INNER, 0, 1);
    }
    const int pbase = t << 4;
    if (pbase + 16 <= N)
      refine_tile<true >(feature, x2, st2, bf, brA, brB, pbase, N, u, sA, qA, sB, qB);
    else
      refine_tile<false>(feature, x2, st2, bf, brA, brB, pbase, N, u, sA, qA, sB, qB);
  }
  atomicAdd(&lsum[cA], sA); atomicAdd(&lsq[cA], qA);
  if (cB < C_IN) { atomicAdd(&lsum[cB], sB); atomicAdd(&lsq[cB], qB); }
  __syncthreads();
  if (threadIdx.x < C_IN) {
    atomicAdd(&stats3[threadIdx.x], lsum[threadIdx.x]);
    atomicAdd(&stats3[C_IN + threadIdx.x], lsq[threadIdx.x]);
  }
}

// ============ kernel 5: out = relu(bn3(u)) @ Wr2^T + br2 ============
template<bool FULL>
__device__ __forceinline__ void out_tile(
    const float* __restrict__ u, const float* __restrict__ st3,
    const v16h (&bo)[2], float brA, float brB,
    int pbase, int N, float* __restrict__ out)
{
  const int lane = threadIdx.x & 31;
  const int half = lane >> 4;
  const int cA = lane & 15, cB = 16 + (lane & 15);
  int p = pbase + (lane & 15);
  if (!FULL && p >= N) p = N - 1;
  const Row20 ru = *(const Row20*)(u + (long long)p * C_IN);
  float z[C_IN];
#pragma unroll
  for (int c = 0; c < C_IN; ++c) z[c] = fmaxf(ru.f[c] * st3[c] + st3[20 + c], 0.f);
  AFrag a;
#pragma unroll
  for (int i = 0; i < 16; ++i) {
    const int K = akmap(i, half);
    a.h[i] = (K < C_IN) ? (_Float16)z[K] : (_Float16)0.f;
  }
  const v8f zz = {};
  DFrag dA, dB;
  dA.v = __builtin_amdgcn_wmma_f32_16x16x32_f16(false, a.v, false, bo[0], (short)0, zz, false, false);
  dB.v = __builtin_amdgcn_wmma_f32_16x16x32_f16(false, a.v, false, bo[1], (short)0, zz, false, false);
#pragma unroll
  for (int v = 0; v < 8; ++v) {
    const int pm = pbase + v + 8 * half;
    if (FULL || pm < N) {
      out[pm * C_IN + cA] = dA.f[v] + brA;
      if (cB < C_IN) out[pm * C_IN + cB] = dB.f[v] + brB;
    }
  }
}

__global__ __launch_bounds__(256) void k_out(
    const float* __restrict__ u, const float* __restrict__ st3,
    const float* __restrict__ Wr2, const float* __restrict__ br2,
    float* __restrict__ out, int N)
{
  const int lane = threadIdx.x & 31;
  const int half = lane >> 4;
  const int cA = lane & 15, cB = 16 + (lane & 15);
  v16h bo[2];
#pragma unroll
  for (int nt = 0; nt < 2; ++nt) {
    AFrag b;
#pragma unroll
    for (int i = 0; i < 16; ++i) {
      const int K = akmap(i, half);
      const int n = (lane & 15) + 16 * nt;
      float v = 0.f;
      if (n < C_IN && K < C_IN) v = Wr2[n * C_IN + K];
      b.h[i] = (_Float16)v;
    }
    bo[nt] = b.v;
  }
  const float brA = br2[cA];
  const float brB = (cB < C_IN) ? br2[cB] : 0.f;
  const int wid = (blockIdx.x * blockDim.x + threadIdx.x) >> 5;
  const int nw  = (gridDim.x * blockDim.x) >> 5;
  const int ntiles = (N + 15) >> 4;
  for (int t = wid; t < ntiles; t += nw) {
    if (t + nw < ntiles) {
      const int pn = ((t + nw) << 4) + (lane & 15);
      __builtin_prefetch(u + (long long)pn * C_IN, 0, 1);
    }
    const int pbase = t << 4;
    if (pbase + 16 <= N)
      out_tile<true >(u, st3, bo, brA, brB, pbase, N, out);
    else
      out_tile<false>(u, st3, bo, brA, brB, pbase, N, out);
  }
}

extern "C" void kernel_launch(void* const* d_in, const int* in_sizes, int n_in,
                              void* d_out, int out_size, void* d_ws, size_t ws_size,
                              hipStream_t stream) {
  const float* points  = (const float*)d_in[0];
  const float* feature = (const float*)d_in[1];
  const int*   index   = (const int*)  d_in[2];
  const float* W1  = (const float*)d_in[3];
  const float* g1  = (const float*)d_in[4];
  const float* b1  = (const float*)d_in[5];
  const float* g2  = (const float*)d_in[6];
  const float* b2  = (const float*)d_in[7];
  const float* Wr1 = (const float*)d_in[8];
  const float* br1 = (const float*)d_in[9];
  const float* g3  = (const float*)d_in[10];
  const float* b3  = (const float*)d_in[11];
  const float* Wr2 = (const float*)d_in[12];
  const float* br2 = (const float*)d_in[13];
  const int N = in_sizes[0] / 3;

  float* ws   = (float*)d_ws;
  float* S1   = ws + 0;     // sum[20], sumsq[20]
  float* S2   = ws + 64;
  float* S3   = ws + 128;
  float* ST1  = ws + 192;   // scale[20], shift[20]
  float* ST2  = ws + 256;
  float* ST3  = ws + 320;
  float* wbuf = ws + 512;                                   // [N, 9]
  const size_t wpad = ((size_t)N * KNN + 3) & ~(size_t)3;   // keep rows 16B aligned
  float* x1   = wbuf + wpad;                                // [N, 20]
  float* x2   = x1   + (size_t)N * C_INNER;                 // [N, 20]
  float* ubuf = x2   + (size_t)N * C_INNER;                 // [N, 20]
  float* out  = (float*)d_out;

  const int ntiles = (N + 15) / 16;               // 16 points per wave-tile
  int wblocks = (ntiles + 7) / 8;                 // 8 waves / 256-thread block
  if (wblocks < 1) wblocks = 1;
  int pblocks = (N + 255) / 256;
  if (pblocks < 1) pblocks = 1;

  k_zero<<<2, 256, 0, stream>>>(ws);
  k_conv_stats<<<wblocks, 256, 0, stream>>>(points, feature, index, W1, S1, N);
  k_finalize<<<1, 32, 0, stream>>>(S1, g1, b1, (float)N * (float)KNN, ST1);
  k_conv_attn<<<wblocks, 256, 0, stream>>>(points, feature, index, W1, ST1, wbuf, x1, N);
  k_propagate<<<pblocks, 256, 0, stream>>>(index, wbuf, x1, x2, S2, N);
  k_finalize<<<1, 32, 0, stream>>>(S2, g2, b2, (float)N, ST2);
  k_refine1<<<wblocks, 256, 0, stream>>>(feature, x2, ST2, Wr1, br1, ubuf, S3, N);
  k_finalize<<<1, 32, 0, stream>>>(S3, g3, b3, (float)N, ST3);
  k_out<<<wblocks, 256, 0, stream>>>(ubuf, ST3, Wr2, br2, out, N);
  (void)n_in; (void)out_size; (void)ws_size; (void)b2; (void)b3;
}
